// EGNN_44358422233330
// MI455X (gfx1250) — matrix operand order repaired and checked by reading, not compiled, not run on hardware
//
#include <hip/hip_runtime.h>

typedef __attribute__((ext_vector_type(8)))  _Float16 v8h;
typedef __attribute__((ext_vector_type(16))) _Float16 v16h;
typedef __attribute__((ext_vector_type(8)))  float    v8f;

#define NB   2
#define NN   768
#define DIMF 64
#define MD   16
#define EH   258      // edge hidden = 2*129
#define KP   288      // padded K for WMMA (9 * 32)
#define HSTR 296      // sH row stride in f16 (592B: 16B aligned, bank-friendly)
#define NH   128      // node hidden

__device__ __forceinline__ float sigmf(float x) { return 1.0f / (1.0f + __expf(-x)); }
__device__ __forceinline__ float siluf(float x) { return x * sigmf(x); }

// Kernel 1: per-node precompute.
//   A[row][k] = be1[k] + sum_d feats[row][d] * We1[d][k]          (i-part + bias)
//   B[row][k] =          sum_d feats[row][d] * We1[64+d][k]       (j-part)
__global__ __launch_bounds__(128) void egnn_pre(
    const float* __restrict__ feats, const float* __restrict__ We1,
    const float* __restrict__ be1,
    float* __restrict__ A, float* __restrict__ Bp)
{
    __shared__ float sF[DIMF];
    const int row = blockIdx.x;
    const int tid = threadIdx.x;
    if (tid < DIMF) sF[tid] = feats[row * DIMF + tid];
    __syncthreads();
    for (int k = tid; k < EH; k += 128) {
        float a = be1[k], b = 0.0f;
        #pragma unroll 8
        for (int d = 0; d < DIMF; ++d) {
            a += sF[d] * We1[d * EH + k];
            b += sF[d] * We1[(DIMF + d) * EH + k];
        }
        A[(size_t)row * EH + k]  = a;
        Bp[(size_t)row * EH + k] = b;
    }
}

// Kernel 2: one workgroup per (b, i). 4 waves, each wave owns whole j-tiles.
__global__ __launch_bounds__(128) void egnn_main(
    const float* __restrict__ feats, const float* __restrict__ coors,
    const float* __restrict__ We1,  const float* __restrict__ We2,
    const float* __restrict__ be2,  const float* __restrict__ Wg,
    const float* __restrict__ bg,   const float* __restrict__ Wn1,
    const float* __restrict__ bn1,  const float* __restrict__ Wn2,
    const float* __restrict__ bn2,
    const float* __restrict__ A,    const float* __restrict__ Bp,
    float* __restrict__ out)
{
    __shared__ __attribute__((aligned(16))) _Float16 sWe2[KP][MD];       // 9216 B
    __shared__ __attribute__((aligned(16))) _Float16 sH[4][MD][HSTR];    // 37888 B
    __shared__ float sA[EH];
    __shared__ float sFeat[DIMF];
    __shared__ float sAcc[4][MD];
    __shared__ float sMi[MD];
    __shared__ float sHn[NH];

    const int row  = blockIdx.x;           // b*NN + i
    const int bb   = row / NN;
    const int tid  = threadIdx.x;
    const int wave = tid >> 5;
    const int lane = tid & 31;

    // Stage We2 as f16, zero-padded to K=288 rows.
    for (int idx = tid; idx < KP * MD; idx += 128) {
        const int k = idx >> 4, n = idx & 15;
        sWe2[k][n] = (k < EH) ? (_Float16)We2[k * MD + n] : (_Float16)0.0f;
    }
    for (int k = tid; k < EH; k += 128) sA[k] = A[(size_t)row * EH + k];
    if (tid < DIMF) sFeat[tid] = feats[row * DIMF + tid];
    __syncthreads();

    const float cx = coors[row * 3 + 0];
    const float cy = coors[row * 3 + 1];
    const float cz = coors[row * 3 + 2];
    const float* __restrict__ Wd = We1 + 128 * EH;   // dist^2 row of We1
    const int   nl   = lane & 15;
    const float be2v = be2[nl];
    const float wgv  = Wg[nl];
    const float bgv  = bg[0];
    const int   khi  = (lane >> 4) << 3;             // A-operand K sub-offset (0|8)

    float acc = 0.0f;                                // m_i partial for N = lane&15

    for (int jt = wave; jt < NN / MD; jt += 4) {
        const int j0 = jt * MD;

        // dist^2 for this tile's 16 j's (held by lanes 0..15)
        float d2l;
        {
            const int jj = j0 + nl;
            const float dx = cx - coors[(bb * NN + jj) * 3 + 0];
            const float dy = cy - coors[(bb * NN + jj) * 3 + 1];
            const float dz = cz - coors[(bb * NN + jj) * 3 + 2];
            d2l = dx * dx + dy * dy + dz * dz;
        }

        // Build silu(h) tile (16 j-rows x 258, padded to 288) in LDS as f16.
        for (int j = 0; j < MD; ++j) {
            const float d2 = __shfl(d2l, j, 32);
            const float* __restrict__ Br = Bp + (size_t)(bb * NN + j0 + j) * EH;
            for (int k = lane; k < EH; k += 32) {
                const float x = sA[k] + Br[k] + d2 * Wd[k];
                sH[wave][j][k] = (_Float16)siluf(x);
            }
            if (lane < (KP - EH)) sH[wave][j][EH + lane] = (_Float16)0.0f;
        }
        // cross-lane LDS RAW hazard before WMMA operand loads (same wave, no barrier)
        asm volatile("s_wait_dscnt 0" ::: "memory");

        // m_pre = silu(h) @ We2 + be2 via 9x v_wmma_f32_16x16x32_f16
        v8f c;
        #pragma unroll
        for (int v = 0; v < 8; ++v) c[v] = be2v;
        #pragma unroll
        for (int s = 0; s < KP / 32; ++s) {
            const _Float16* ap = &sH[wave][nl][s * 32 + khi];
            const v8h a0 = *(const v8h*)ap;
            const v8h a1 = *(const v8h*)(ap + 16);
            const v16h aa = __builtin_shufflevector(a0, a1,
                0,1,2,3,4,5,6,7,8,9,10,11,12,13,14,15);
            v16h bbv;
            #pragma unroll
            for (int e = 0; e < 16; ++e)
                bbv[e] = sWe2[s * 32 + khi + (e & 7) + ((e >> 3) << 4)][nl];
            c = __builtin_amdgcn_wmma_f32_16x16x32_f16(
                    false, aa, false, bbv, (short)0, c, false, false);
        }

        // m = silu(m_pre); gate = sigmoid(m . Wg + bg); acc += m * gate (sum over j)
        float mv[8], pv[8];
        #pragma unroll
        for (int v = 0; v < 8; ++v) { mv[v] = siluf(c[v]); pv[v] = mv[v] * wgv; }
        #pragma unroll
        for (int off = 1; off <= 8; off <<= 1) {
            #pragma unroll
            for (int v = 0; v < 8; ++v) pv[v] += __shfl_xor(pv[v], off, 32);
        }
        #pragma unroll
        for (int v = 0; v < 8; ++v) acc += mv[v] * sigmf(pv[v] + bgv);
    }

    acc += __shfl_xor(acc, 16, 32);                  // combine half-wave row groups
    if (lane < MD) sAcc[wave][lane] = acc;
    __syncthreads();
    if (tid < MD) sMi[tid] = sAcc[0][tid] + sAcc[1][tid] + sAcc[2][tid] + sAcc[3][tid];
    __syncthreads();

    // Node MLP: [feats(64) | m_i(16)] -> 128 (silu) -> 64, + residual
    {
        float h = bn1[tid];
        #pragma unroll 8
        for (int d = 0; d < DIMF; ++d) h += sFeat[d] * Wn1[d * NH + tid];
        #pragma unroll
        for (int m = 0; m < MD; ++m)  h += sMi[m] * Wn1[(DIMF + m) * NH + tid];
        sHn[tid] = siluf(h);
    }
    __syncthreads();
    if (tid < DIMF) {
        float o = bn2[tid] + sFeat[tid];
        #pragma unroll 8
        for (int hh = 0; hh < NH; ++hh) o += sHn[hh] * Wn2[hh * DIMF + tid];
        out[(size_t)row * DIMF + tid] = o;
    }
    // second tuple element: coors pass-through
    if (tid < 3) out[(size_t)NB * NN * DIMF + (size_t)row * 3 + tid] = coors[row * 3 + tid];
}

extern "C" void kernel_launch(void* const* d_in, const int* in_sizes, int n_in,
                              void* d_out, int out_size, void* d_ws, size_t ws_size,
                              hipStream_t stream) {
    (void)in_sizes; (void)n_in; (void)out_size; (void)ws_size;
    const float* feats = (const float*)d_in[0];
    const float* coors = (const float*)d_in[1];
    const float* We1   = (const float*)d_in[2];
    const float* be1   = (const float*)d_in[3];
    const float* We2   = (const float*)d_in[4];
    const float* be2   = (const float*)d_in[5];
    const float* Wg    = (const float*)d_in[6];
    const float* bg    = (const float*)d_in[7];
    const float* Wn1   = (const float*)d_in[8];
    const float* bn1   = (const float*)d_in[9];
    const float* Wn2   = (const float*)d_in[10];
    const float* bn2   = (const float*)d_in[11];
    float* out = (float*)d_out;

    float* A  = (float*)d_ws;                    // NB*NN*EH floats
    float* Bp = A + (size_t)NB * NN * EH;        // NB*NN*EH floats  (~3.2 MB total)

    egnn_pre<<<NB * NN, 128, 0, stream>>>(feats, We1, be1, A, Bp);
    egnn_main<<<NB * NN, 128, 0, stream>>>(feats, coors, We1, We2, be2, Wg, bg,
                                           Wn1, bn1, Wn2, bn2, A, Bp, out);
}
